// SDPAWithRingBufferBoolMask_36017595744259
// MI455X (gfx1250) — matrix operand order repaired and checked by reading, not compiled, not run on hardware
//
#include <hip/hip_runtime.h>
#include <math.h>

#define WINDOW 2048
#define BUFSZ  4096
#define SEQ    1024
#define DIM    64

typedef _Float16     v16h __attribute__((ext_vector_type(16)));
typedef float        v8f  __attribute__((ext_vector_type(8)));
typedef float        v4f  __attribute__((ext_vector_type(4)));
typedef unsigned int v4u  __attribute__((ext_vector_type(4)));

union U16h { v4u u[2]; v16h h; };

__device__ __forceinline__ unsigned pk2(float a, float b) {
  union { _Float16 hh[2]; unsigned u; } x;
  x.hh[0] = (_Float16)a; x.hh[1] = (_Float16)b;
  return x.u;
}

__device__ __forceinline__ v8f wmma16(v16h a, v16h b, v8f c) {
  // D = A(16x32 f16) * B(32x16 f16) + C(16x16 f32)
  return __builtin_amdgcn_wmma_f32_16x16x32_f16(false, a, false, b, (short)0, c,
                                                false, false);
}

// padded LDS row strides (f16 elements) to avoid bank conflicts on b128 accesses
#define QROW 72
#define KROW 72
#define VROW 40
#define VT_OFF 4608  // bytes = 32*KROW*2

__global__ __launch_bounds__(128) void sdpa_ring_kernel(
    const float* __restrict__ gq, const float* __restrict__ gk,
    const float* __restrict__ gv, const int* __restrict__ gsp,
    float* __restrict__ gout) {
  __shared__ __align__(16) unsigned char smem[9728];
  _Float16* Qs = (_Float16*)smem;             // [64][QROW]  (overlaid with tiles)
  _Float16* Kt = (_Float16*)smem;             // [32][KROW]  key rows, f16
  _Float16* Vt = (_Float16*)(smem + VT_OFF);  // [64][VROW]  V transposed, f16

  const int t    = threadIdx.x;
  const int lane = t & 31;
  const int w    = t >> 5;      // wave id, 4 waves
  const int hh   = lane >> 4;   // lane half
  const int nn   = lane & 15;   // query col (softmax) / d col (O)

  const int bid = blockIdx.x;   // 1024 blocks = 64 (b*h) x 16 q-tiles
  const int bh  = bid >> 4;
  const int q0  = (bid & 15) * 64;

  const int start = gsp[0];
  const int T     = start + SEQ;  // total_written

  const float* qb = gq + (size_t)bh * SEQ * DIM;
  const float* kb = gk + (size_t)bh * BUFSZ * DIM;
  const float* vb = gv + (size_t)bh * BUFSZ * DIM;

  // ---- stage Q tile (64 x 64), pre-scaled by 1/sqrt(D)=0.125, as f16 ----
  {
    const int qr = t >> 1;
    const int sg = (t & 1) * 32;
    const v4f* src = (const v4f*)(qb + (size_t)(q0 + qr) * DIM + sg);
    unsigned* dst  = (unsigned*)(Qs + qr * QROW + sg);
#pragma unroll
    for (int i = 0; i < 8; ++i) {
      v4f f = src[i];
      dst[2 * i]     = pk2(f[0] * 0.125f, f[1] * 0.125f);
      dst[2 * i + 1] = pk2(f[2] * 0.125f, f[3] * 0.125f);
    }
  }
  __syncthreads();

  // ---- per-wave Q^T B-operands (d=0..31 and d=32..63); live in regs all kernel ----
  const int qw = q0 + w * 16;
  U16h qB0, qB1;
  {
    const v4u* r4 = (const v4u*)(Qs + (w * 16 + nn) * QROW);
    qB0.u[0] = r4[hh];     qB0.u[1] = r4[2 + hh];
    qB1.u[0] = r4[4 + hh]; qB1.u[1] = r4[6 + hh];
  }

  const float NEGINF = -__builtin_inff();
  float m = NEGINF, l = 0.0f;
  v8f Oc[4] = {};  // O: d-tile dt, reg r -> (query = qw + r + 8*hh, d = dt*16 + nn)

  const int P0 = start + q0;        // lowest absolute query pos in block
  const int Pw = start + qw;        // lowest absolute query pos in wave
  int lo = P0 - (WINDOW - 1); if (lo < 0) lo = 0;
  const int kfirst = lo & ~31;
  const int klast  = P0 + 63;       // == max p <= T-1, so always in-bounds

  for (int kc = kfirst; kc <= klast; kc += 32) {
    __syncthreads();
    // ---- K chunk -> LDS f16 rows [32][KROW] ----
    {
      const int row = t >> 2, sg = (t & 3) * 16;
      const v4f* src = (const v4f*)(kb + (size_t)(kc + row) * DIM + sg);
      unsigned* dst  = (unsigned*)(Kt + row * KROW + sg);
#pragma unroll
      for (int i = 0; i < 4; ++i) {
        v4f f = src[i];
        dst[2 * i]     = pk2(f[0], f[1]);
        dst[2 * i + 1] = pk2(f[2], f[3]);
      }
    }
    // ---- V chunk -> LDS f16 transposed [64 d][VROW keys], paired b32 stores ----
    {
      const int kp = (t & 15) * 2;   // this thread owns keys kp, kp+1
      const int sg = (t >> 4) * 8;   // and 8 d-values sg..sg+7
      const float* r0 = vb + (size_t)(kc + kp) * DIM + sg;
      const float* r1 = r0 + DIM;
      v4f a0 = *(const v4f*)(r0);
      v4f a1 = *(const v4f*)(r0 + 4);
      v4f b0 = *(const v4f*)(r1);
      v4f b1 = *(const v4f*)(r1 + 4);
#pragma unroll
      for (int i = 0; i < 4; ++i) {
        *(unsigned*)(Vt + (sg + i) * VROW + kp)     = pk2(a0[i], b0[i]);
        *(unsigned*)(Vt + (sg + 4 + i) * VROW + kp) = pk2(a1[i], b1[i]);
      }
    }
    // prefetch next chunk into cache hierarchy
    if (kc + 32 <= klast) {
      const int row = t >> 2, sg = (t & 3) * 16;
      __builtin_prefetch(kb + (size_t)(kc + 32 + row) * DIM + sg, 0, 1);
      __builtin_prefetch(vb + (size_t)(kc + 32 + row) * DIM + sg, 0, 1);
    }
    __syncthreads();

    // per-wave uniform skip of fully-invalid chunks
    if (kc > Pw + 15) continue;                // all keys in the future
    if (Pw - (kc + 31) >= WINDOW) continue;    // all keys outside window

    // ---- S^T = K * Q^T : two 16-key C-tiles, K-dim = D in two halves ----
    v8f c0 = {}, c1 = {};
    {
      U16h a;
      const v4u* r0 = (const v4u*)(Kt + nn * KROW);
      const v4u* r1 = (const v4u*)(Kt + (16 + nn) * KROW);
      a.u[0] = r0[hh];     a.u[1] = r0[2 + hh]; c0 = wmma16(a.h, qB0.h, c0);
      a.u[0] = r0[4 + hh]; a.u[1] = r0[6 + hh]; c0 = wmma16(a.h, qB1.h, c0);
      a.u[0] = r1[hh];     a.u[1] = r1[2 + hh]; c1 = wmma16(a.h, qB0.h, c1);
      a.u[0] = r1[4 + hh]; a.u[1] = r1[6 + hh]; c1 = wmma16(a.h, qB1.h, c1);
    }

    // ---- ring-buffer mask on edge chunks only ----
    const bool interior = (kc + 31 <= Pw) && (Pw + 15 - kc < WINDOW);
    if (!interior) {
      const int p = Pw + nn;  // absolute query position of this lane's column
#pragma unroll
      for (int r = 0; r < 8; ++r) {
        const int j0  = kc + 8 * hh + r;
        const int cp0 = j0 + (((T - 1 - j0) >> 12) << 12);  // floor-div wrap
        const int d0  = p - cp0;
        if (cp0 < 0 || d0 < 0 || d0 >= WINDOW) c0[r] = NEGINF;
        const int j1  = j0 + 16;
        const int cp1 = j1 + (((T - 1 - j1) >> 12) << 12);
        const int d1  = p - cp1;
        if (cp1 < 0 || d1 < 0 || d1 >= WINDOW) c1[r] = NEGINF;
      }
    }

    // ---- online softmax: stats per-lane (query = lane&15), one cross-half shuffle ----
    float mc = NEGINF;
#pragma unroll
    for (int r = 0; r < 8; ++r) { mc = fmaxf(mc, c0[r]); mc = fmaxf(mc, c1[r]); }
    mc = fmaxf(mc, __shfl_xor(mc, 16, 32));
    const float mn    = fmaxf(m, mc);
    const float mns   = (mn == NEGINF) ? 0.0f : mn;  // -inf-safe
    const float alpha = __expf(m - mns);
    float p0[8], p1[8];
    float rs = 0.0f;
#pragma unroll
    for (int r = 0; r < 8; ++r) {
      p0[r] = __expf(c0[r] - mns);
      p1[r] = __expf(c1[r] - mns);
      rs += p0[r] + p1[r];
    }
    rs += __shfl_xor(rs, 16, 32);
    l = l * alpha + rs;
    m = mn;

    // ---- P as A-operand: exact in-lane match with S^T C-layout (no shuffles) ----
    U16h pa;
#pragma unroll
    for (int r = 0; r < 8; ++r) {
      pa.h[r]     = (_Float16)p0[r];
      pa.h[8 + r] = (_Float16)p1[r];
    }

    // ---- broadcast per-query alpha into the O row dimension ----
    v8f alphaV;
#pragma unroll
    for (int r = 0; r < 8; ++r) alphaV[r] = __shfl(alpha, r + 8 * hh, 32);

    // ---- O = alphaV*O + P * V  (4 d-tiles, K-dim = 32 keys) ----
#pragma unroll
    for (int dt = 0; dt < 4; ++dt) {
      U16h vB;
      const v4u* r4 = (const v4u*)(Vt + (dt * 16 + nn) * VROW);
      vB.u[0] = r4[hh]; vB.u[1] = r4[2 + hh];
      Oc[dt] = wmma16(pa.h, vB.h, Oc[dt] * alphaV);
    }
  }

  // ---- epilogue: divide by per-query l (broadcast), coalesced stores ----
  const float invl = 1.0f / l;
  float invlR[8];
#pragma unroll
  for (int r = 0; r < 8; ++r) invlR[r] = __shfl(invl, r + 8 * hh, 32);
  float* obase = gout + (size_t)(bh * SEQ + qw + 8 * hh) * DIM + nn;
#pragma unroll
  for (int dt = 0; dt < 4; ++dt) {
#pragma unroll
    for (int r = 0; r < 8; ++r)
      obase[(size_t)r * DIM + dt * 16] = Oc[dt][r] * invlR[r];
  }
}

extern "C" void kernel_launch(void* const* d_in, const int* in_sizes, int n_in,
                              void* d_out, int out_size, void* d_ws, size_t ws_size,
                              hipStream_t stream) {
  const float* q  = (const float*)d_in[0];
  const float* k  = (const float*)d_in[1];
  const float* v  = (const float*)d_in[2];
  const int*   sp = (const int*)d_in[3];
  float* out = (float*)d_out;
  (void)in_sizes; (void)n_in; (void)out_size; (void)d_ws; (void)ws_size;
  dim3 grid(4 * 16 * (SEQ / 64));  // 1024 workgroups
  sdpa_ring_kernel<<<grid, 128, 0, stream>>>(q, k, v, sp, out);
}
